// NanoMamba_13941463843635
// MI455X (gfx1250) — matrix operand order repaired and checked by reading, not compiled, not run on hardware
//
#include <hip/hip_runtime.h>
#include <hip/hip_bf16.h>
#include <math.h>

// ---- problem dims ----
#define L_LAYERS 2
#define DM 256          // d_model
#define DI 384          // d_inner
#define NS 16           // d_state
#define KC 3            // conv width
#define NM 40           // n_mels
#define BB 16           // batch
#define TT 512          // seq
#define MTOK (BB*TT)    // 8192 tokens

typedef __bf16        v16bf __attribute__((ext_vector_type(16)));
typedef float         v8f   __attribute__((ext_vector_type(8)));
typedef unsigned int  v4u   __attribute__((ext_vector_type(4)));

__device__ __forceinline__ unsigned short f2bf(float f) {
  union { float f; unsigned u; } v; v.f = f;
  unsigned r = v.u + 0x7FFFu + ((v.u >> 16) & 1u);   // round-to-nearest-even
  return (unsigned short)(r >> 16);
}
__device__ __forceinline__ float sigmoidf_(float x) { return 1.f / (1.f + __expf(-x)); }
__device__ __forceinline__ float siluf_(float x)    { return x * sigmoidf_(x); }
__device__ __forceinline__ float softplusf_(float x) {
  if (x >  20.f) return x;
  if (x < -20.f) return __expf(x);
  return log1pf(__expf(x));
}

// ---------------- fp32 -> bf16 weight conversion ----------------
__global__ void cvt_bf16_kernel(const float* __restrict__ src,
                                unsigned short* __restrict__ dst, int n) {
  int i = blockIdx.x * blockDim.x + threadIdx.x;
  if (i < n) dst[i] = f2bf(src[i]);
}

// ---------------- LayerNorm, one wave32 per token, emits bf16 ----------------
__global__ void ln_kernel(const float* __restrict__ x,
                          const float* __restrict__ w,
                          const float* __restrict__ bia,
                          unsigned short* __restrict__ hbf) {
  int wave = threadIdx.x >> 5, lane = threadIdx.x & 31;
  int m = blockIdx.x * 8 + wave;               // grid sized exactly: MTOK/8 blocks
  const float* row = x + (size_t)m * DM;
  float v[8]; float s = 0.f;
  #pragma unroll
  for (int i = 0; i < 8; ++i) { v[i] = row[lane * 8 + i]; s += v[i]; }
  #pragma unroll
  for (int off = 16; off >= 1; off >>= 1) s += __shfl_xor(s, off, 32);
  float mu = s * (1.f / DM);
  float var = 0.f;
  #pragma unroll
  for (int i = 0; i < 8; ++i) { float dd = v[i] - mu; var += dd * dd; }
  #pragma unroll
  for (int off = 16; off >= 1; off >>= 1) var += __shfl_xor(var, off, 32);
  float rs = rsqrtf(var * (1.f / DM) + 1e-5f);
  #pragma unroll
  for (int i = 0; i < 8; ++i) {
    int c = lane * 8 + i;
    hbf[(size_t)m * DM + c] = f2bf((v[i] - mu) * rs * w[c] + bia[c]);
  }
}

// ---------------- WMMA bf16 GEMM: C[M,N] = A[M,K] * W[N,K]^T (+resid) ----------------
// One wave per 16x64 output strip (4 accumulators) -> A-fragment reused 4x.
// K-loop in steps of 32 (v_wmma_f32_16x16x32_bf16). N must be a multiple of 64.
__global__ void wmma_gemm_bf16(const unsigned short* __restrict__ A,
                               const unsigned short* __restrict__ W,
                               const float* __restrict__ resid,
                               float* __restrict__ C,
                               int M, int N, int K) {
  int wave = threadIdx.x >> 5;
  int lane = threadIdx.x & 31;
  int superN = N >> 6;                                // 64-wide column groups
  int tile = blockIdx.x * (blockDim.x >> 5) + wave;
  if (tile >= (M >> 4) * superN) return;
  int tm = tile / superN, tg = tile % superN;
  int r  = lane & 15;
  int hi = lane >> 4;

  union Frag { v16bf v; v4u q[2]; };
  v8f acc0 = {}, acc1 = {}, acc2 = {}, acc3 = {};
  // A layout (16x32): lane = row m, lanes 0-15 hold K 0-15, lanes 16-31 hold K 16-31
  const unsigned short* arow = A + (size_t)(tm * 16 + r) * K + (hi ? 16 : 0);
  // B layout (32x16): lane = column n, lanes 0-15 hold K 0-15, lanes 16-31 hold K 16-31
  const unsigned short* brow = W + (size_t)(tg * 64 + r) * K + (hi ? 16 : 0);
  const size_t bstep = (size_t)16 * K;                // next 16-column tile of W

  for (int kk = 0; kk < K; kk += 32) {
    Frag a, b0, b1, b2, b3;
    const v4u* pa = reinterpret_cast<const v4u*>(arow + kk);
    a.q[0] = pa[0]; a.q[1] = pa[1];
    const v4u* pb0 = reinterpret_cast<const v4u*>(brow + kk);
    const v4u* pb1 = reinterpret_cast<const v4u*>(brow + bstep + kk);
    const v4u* pb2 = reinterpret_cast<const v4u*>(brow + 2 * bstep + kk);
    const v4u* pb3 = reinterpret_cast<const v4u*>(brow + 3 * bstep + kk);
    b0.q[0] = pb0[0]; b0.q[1] = pb0[1];
    b1.q[0] = pb1[0]; b1.q[1] = pb1[1];
    b2.q[0] = pb2[0]; b2.q[1] = pb2[1];
    b3.q[0] = pb3[0]; b3.q[1] = pb3[1];
    __builtin_prefetch(arow + kk + 64, 0, 1);         // global_prefetch_b8
    __builtin_prefetch(brow + kk + 64, 0, 1);
    acc0 = __builtin_amdgcn_wmma_f32_16x16x32_bf16(false, a.v, false, b0.v, (short)0, acc0, false, false);
    acc1 = __builtin_amdgcn_wmma_f32_16x16x32_bf16(false, a.v, false, b1.v, (short)0, acc1, false, false);
    acc2 = __builtin_amdgcn_wmma_f32_16x16x32_bf16(false, a.v, false, b2.v, (short)0, acc2, false, false);
    acc3 = __builtin_amdgcn_wmma_f32_16x16x32_bf16(false, a.v, false, b3.v, (short)0, acc3, false, false);
  }
  // C/D layout: lane<16 -> M = i, N = lane ; lane>=16 -> M = 8+i, N = lane-16
  int mbase = tm * 16 + (hi ? 8 : 0);
  v8f accs[4] = {acc0, acc1, acc2, acc3};
  #pragma unroll
  for (int j = 0; j < 4; ++j) {
    int n = tg * 64 + j * 16 + r;
    #pragma unroll
    for (int i = 0; i < 8; ++i) {
      size_t idx = (size_t)(mbase + i) * N + n;
      float v = accs[j][i];
      if (resid) v += resid[idx];
      C[idx] = v;
    }
  }
}

// ---------------- depthwise causal conv (K=3) + SiLU ----------------
__global__ void conv_silu_kernel(const float* __restrict__ xz,   // [M, 2*DI], xb = cols [0,DI)
                                 const float* __restrict__ cw,   // [DI, KC]
                                 const float* __restrict__ cb,   // [DI]
                                 float* __restrict__ xb) {       // [M, DI]
  int idx = blockIdx.x * blockDim.x + threadIdx.x;
  if (idx >= BB * TT * DI) return;
  int d = idx % DI;
  int t = (idx / DI) % TT;
  int b = idx / (DI * TT);
  const float* col = xz + ((size_t)b * TT) * (2 * DI) + d;
  float acc = cb[d];
  #pragma unroll
  for (int k = 0; k < KC; ++k) {
    int tt = t - (KC - 1) + k;
    if (tt >= 0) acc += cw[d * KC + k] * col[(size_t)tt * (2 * DI)];
  }
  xb[idx] = siluf_(acc);
}

// ---------------- small projections + SNR gating, one block per token ----------------
__global__ void xproj_kernel(const float* __restrict__ xb,     // [M, DI]
                             const float* __restrict__ snr,    // [M, NM]
                             const float* __restrict__ xw,     // [2N+1, DI] = 33 x 384
                             const float* __restrict__ sw,     // [N+1, NM]  = 17 x 40
                             const float* __restrict__ sb,     // [17]
                             const float* __restrict__ alpha_p,
                             const float* __restrict__ gf_p,
                             float* __restrict__ dts,          // [M]
                             float* __restrict__ Beff,         // [M, NS]
                             float* __restrict__ Cb) {         // [M, NS]
  __shared__ float sm[64];
  int m = blockIdx.x;
  int p = threadIdx.x;
  if (p < 33) {
    const float* xr = xb + (size_t)m * DI;
    const float* wr = xw + p * DI;
    float s = 0.f;
    for (int k = 0; k < DI; ++k) s += xr[k] * wr[k];
    sm[p] = s;
  } else if (p < 50) {
    int q = p - 33;
    const float* xr = snr + (size_t)m * NM;
    const float* wr = sw + q * NM;
    float s = sb[q];
    for (int k = 0; k < NM; ++k) s += xr[k] * wr[k];
    sm[p] = s;
  }
  __syncthreads();
  float alpha = alpha_p[0], gf = gf_p[0];
  if (p == 0) dts[m] = sm[0] + sm[33];                    // dt_raw + dt_shift
  if (p >= 1 && p <= 16) {
    int n = p - 1;
    float gate = gf + (1.f - gf) * sigmoidf_(sm[34 + n]); // B_gate
    Beff[(size_t)m * NS + n] = sm[1 + n] * (1.f - alpha + alpha * gate);
  }
  if (p >= 17 && p <= 32) {
    Cb[(size_t)m * NS + (p - 17)] = sm[p];
  }
}

// ---------------- selective scan: grid (B, DI/128), 128 threads, 1 thread/channel ---------
#define TCH 64
#define SCB 128   // threads per scan block
__global__ void scan_kernel(const float* __restrict__ xb,      // [M, DI]
                            const float* __restrict__ xz,      // [M, 2*DI], z = cols [DI,2DI)
                            const float* __restrict__ dts,     // [M]
                            const float* __restrict__ Beff,    // [M, NS]
                            const float* __restrict__ Cb,      // [M, NS]
                            const float* __restrict__ dtw_p,   // [DI]
                            const float* __restrict__ dtb_p,   // [DI]
                            const float* __restrict__ Alog,    // [DI, NS]
                            const float* __restrict__ Dskip,   // [DI]
                            unsigned short* __restrict__ ybf)  // [M, DI] bf16
{
  __shared__ float sB[TCH][NS];
  __shared__ float sC[TCH][NS];
  __shared__ float sD[TCH];
  int b = blockIdx.x;
  int d = blockIdx.y * SCB + threadIdx.x;
  int tid = threadIdx.x;
  float dtw = dtw_p[d], dtb = dtb_p[d], dsk = Dskip[d];
  float Ad[NS], h[NS];
  #pragma unroll
  for (int n = 0; n < NS; ++n) { Ad[n] = -__expf(Alog[d * NS + n]); h[n] = 0.f; }

  for (int t0 = 0; t0 < TT; t0 += TCH) {
    __syncthreads();
    for (int idx = tid; idx < TCH * NS; idx += SCB) {
      int i = idx >> 4, n = idx & 15;
      size_t g = ((size_t)b * TT + t0 + i) * NS + n;
      sB[i][n] = Beff[g];
      sC[i][n] = Cb[g];
    }
    for (int idx = tid; idx < TCH; idx += SCB)
      sD[idx] = dts[(size_t)b * TT + t0 + idx];
    __syncthreads();

    for (int i = 0; i < TCH; ++i) {
      size_t m = (size_t)b * TT + t0 + i;
      float dtv = softplusf_(sD[i] * dtw + dtb);
      float xv  = xb[m * DI + d];
      float dx  = dtv * xv;
      float y = 0.f;
      #pragma unroll
      for (int n = 0; n < NS; ++n) {
        float dA = __expf(Ad[n] * dtv);
        h[n] = dA * h[n] + dx * sB[i][n];
        y += h[n] * sC[i][n];
      }
      y += dsk * xv;
      float zv = xz[m * (2 * DI) + DI + d];
      ybf[m * DI + d] = f2bf(y * siluf_(zv));   // fused z-gate + bf16 for outproj WMMA
    }
  }
}

// ---------------- host orchestration ----------------
extern "C" void kernel_launch(void* const* d_in, const int* in_sizes, int n_in,
                              void* d_out, int out_size, void* d_ws, size_t ws_size,
                              hipStream_t stream) {
  (void)in_sizes; (void)n_in; (void)out_size; (void)ws_size;
  const float* x_in   = (const float*)d_in[0];
  const float* snr    = (const float*)d_in[1];
  const float* norm_w = (const float*)d_in[2];
  const float* norm_b = (const float*)d_in[3];
  const float* inpw   = (const float*)d_in[4];   // [L, 2*DI, DM]
  const float* convw  = (const float*)d_in[5];   // [L, DI, 1, KC]
  const float* convb  = (const float*)d_in[6];
  const float* xprojw = (const float*)d_in[7];   // [L, 33, DI]
  const float* snrw   = (const float*)d_in[8];   // [L, 17, NM]
  const float* snrb   = (const float*)d_in[9];
  const float* dtw    = (const float*)d_in[10];  // [L, DI, 1]
  const float* dtb    = (const float*)d_in[11];
  const float* alog   = (const float*)d_in[12];  // [L, DI, NS]
  const float* dskip  = (const float*)d_in[13];
  const float* alph   = (const float*)d_in[14];
  const float* gfl    = (const float*)d_in[15];
  const float* outw   = (const float*)d_in[16];  // [L, DM, DI]
  float* out = (float*)d_out;

  char* p = (char*)d_ws;
  auto alloc = [&](size_t bytes) { char* r = p; p += (bytes + 255) & ~(size_t)255; return r; };
  float*          xbuf  = (float*)alloc((size_t)MTOK * DM * 4);
  unsigned short* hbf   = (unsigned short*)alloc((size_t)MTOK * DM * 2);
  float*          xz    = (float*)alloc((size_t)MTOK * 2 * DI * 4);
  float*          xbb   = (float*)alloc((size_t)MTOK * DI * 4);
  float*          dtsum = (float*)alloc((size_t)MTOK * 4);
  float*          Beff  = (float*)alloc((size_t)MTOK * NS * 4);
  float*          Cbuf  = (float*)alloc((size_t)MTOK * NS * 4);
  unsigned short* ybf   = (unsigned short*)alloc((size_t)MTOK * DI * 2);
  unsigned short* wbin  = (unsigned short*)alloc((size_t)2 * DI * DM * 2);
  unsigned short* wbout = (unsigned short*)alloc((size_t)DM * DI * 2);

  hipMemcpyAsync(xbuf, x_in, (size_t)MTOK * DM * 4, hipMemcpyDeviceToDevice, stream);

  for (int l = 0; l < L_LAYERS; ++l) {
    const int nin  = 2 * DI * DM;   // in_proj weights
    const int nout = DM * DI;       // out_proj weights
    cvt_bf16_kernel<<<(nin + 255) / 256, 256, 0, stream>>>(inpw + (size_t)l * nin, wbin, nin);
    cvt_bf16_kernel<<<(nout + 255) / 256, 256, 0, stream>>>(outw + (size_t)l * nout, wbout, nout);

    ln_kernel<<<MTOK / 8, 256, 0, stream>>>(xbuf, norm_w + l * DM, norm_b + l * DM, hbf);

    // xz = hbf * in_proj_w^T   (8192 x 256 -> 768); 16x64 strips per wave
    {
      int tiles = (MTOK / 16) * ((2 * DI) / 64);
      wmma_gemm_bf16<<<(tiles + 7) / 8, 256, 0, stream>>>(hbf, wbin, nullptr, xz, MTOK, 2 * DI, DM);
    }

    conv_silu_kernel<<<(BB * TT * DI + 255) / 256, 256, 0, stream>>>(
        xz, convw + (size_t)l * DI * KC, convb + l * DI, xbb);

    xproj_kernel<<<MTOK, 64, 0, stream>>>(
        xbb, snr, xprojw + (size_t)l * 33 * DI, snrw + (size_t)l * 17 * NM,
        snrb + l * 17, alph + l, gfl + l, dtsum, Beff, Cbuf);

    scan_kernel<<<dim3(BB, DI / SCB), SCB, 0, stream>>>(
        xbb, xz, dtsum, Beff, Cbuf, dtw + (size_t)l * DI, dtb + l * DI,
        alog + (size_t)l * DI * NS, dskip + l * DI, ybf);

    // out = ybf * outproj_w^T + residual   (8192 x 384 -> 256)
    float* dst = (l == L_LAYERS - 1) ? out : xbuf;
    {
      int tiles = (MTOK / 16) * (DM / 64);
      wmma_gemm_bf16<<<(tiles + 7) / 8, 256, 0, stream>>>(ybf, wbout, xbuf, dst, MTOK, DM, DI);
    }
  }
}
